// Parser_51092930953403
// MI455X (gfx1250) — matrix operand order repaired and checked by reading, not compiled, not run on hardware
//
#include <hip/hip_runtime.h>
#include <hip/hip_fp16.h>
#include <stdint.h>

// ---------------- model dims ----------------
#define NB    16      // B
#define ML    64
#define MAL   128
#define ELE   6
#define HH    256     // H
#define INF_  132     // WE+PE
#define G4    1024    // 4*H
#define D2H   512     // 2*H
#define WRS_  1024    // 4*H
#define NCAT  100
#define K1    6144    // ELE*WRS
#define N1    2048    // 2*WRS

typedef __attribute__((ext_vector_type(16))) _Float16 v16h;
typedef __attribute__((ext_vector_type(8)))  float    v8f;

__device__ __forceinline__ float sigmf(float x) { return 1.f / (1.f + __expf(-x)); }

// ---- WMMA fragment builders (16x16x32 f16, contiguous k = 16*(lane/16) + j in A and B) ----
__device__ __forceinline__ v16h load_a_frag(const _Float16* As, int ldA, int mbase, int k0) {
  int lane = threadIdx.x & 31;
  int m  = lane & 15;
  int kb = (lane < 16) ? 0 : 16;
  v16h a;
#pragma unroll
  for (int j = 0; j < 16; ++j) {
    int v = j >> 1;
    int k = j + kb + 0 * v;
    a[j] = As[(mbase + m) * ldA + k0 + k];
  }
  return a;
}
__device__ __forceinline__ v16h load_b_frag(const _Float16* Bs, int ldB, int nbase, int k0) {
  int lane = threadIdx.x & 31;
  int n  = lane & 15;
  int kb = (lane < 16) ? 0 : 16;
  v16h b;
#pragma unroll
  for (int j = 0; j < 16; ++j) b[j] = Bs[(nbase + n) * ldB + k0 + kb + j];
  return b;
}

// ---- async global->LDS copy of one 512B chunk per wave per instruction, 16B per lane (wave32) ----
// Copies `nbytes` (multiple of 512) from src to LDS dst; caller = one wave.
__device__ __forceinline__ void async_copy_wave(const void* src, void* ldsDst, int nbytes) {
  int lane = threadIdx.x & 31;
  unsigned ldsOff = (unsigned)(uintptr_t)ldsDst + (unsigned)(lane * 16);
  unsigned long long ga = (unsigned long long)(uintptr_t)src + (unsigned long long)(lane * 16);
  for (int c = 0; c < nbytes; c += 512) {
    asm volatile("global_load_async_to_lds_b128 %0, %1, off offset:%2"
                 :: "v"(ldsOff), "v"(ga), "i"(c) : "memory");
  }
}
__device__ __forceinline__ void wait_async0() {
  asm volatile("s_wait_asynccnt 0x0" ::: "memory");
}

// ---------------- stage 0: f32 -> f16 weight conversion ----------------
__global__ void cvt16_kernel(const float* __restrict__ src, _Float16* __restrict__ dst, int n) {
  int i = blockIdx.x * 256 + threadIdx.x;
  if (i < n) dst[i] = (_Float16)src[i];
}
__global__ void cvtW2_kernel(const float* __restrict__ w2, _Float16* __restrict__ dst) {
  int i = blockIdx.x * 256 + threadIdx.x;   // 128*2048 total
  int r = i >> 11;
  dst[i] = (_Float16)((r < NCAT) ? w2[i] : 0.f);
}

// ---------------- stage 1: embedding concat -> x (T,B,132) ----------------
__global__ void embed_kernel(const float* __restrict__ wemb, const float* __restrict__ pemb,
                             const int* __restrict__ dic, const int* __restrict__ pos,
                             float* __restrict__ x) {
  int b = blockIdx.x >> 6, t = blockIdx.x & 63;
  int c = threadIdx.x;
  if (c >= INF_) return;
  float v;
  if (c < 100) v = wemb[dic[b * ML + t] * 100 + c];
  else         v = pemb[pos[b * ML + t] * 32 + (c - 100)];
  x[(t * NB + b) * INF_ + c] = v;
}

// ---------------- stage 2: bi-LSTM (fp32 scalar; tiny fraction of FLOPs) ----------------
__global__ void bilstm_kernel(const float* __restrict__ x,
                              const float* WihF, const float* WhhF, const float* bF,
                              const float* WihB, const float* WhhB, const float* bB,
                              const int* __restrict__ lens,
                              float* __restrict__ bi_out, float* __restrict__ out_r) {
  __shared__ float hs[NB][HH];
  __shared__ float cs[NB][HH];
  __shared__ float gs[NB][G4];
  const int dir = blockIdx.x;
  const float* Wih = dir ? WihB : WihF;
  const float* Whh = dir ? WhhB : WhhF;
  const float* bia = dir ? bB   : bF;
  for (int i = threadIdx.x; i < NB * HH; i += 1024) { hs[i >> 8][i & 255] = 0.f; cs[i >> 8][i & 255] = 0.f; }
  __syncthreads();
  const int col = threadIdx.x;              // 0..1023 gate column
  for (int t = 0; t < ML; ++t) {
    for (int row = 0; row < NB; ++row) {
      int len = lens[row];
      int st  = dir ? ((t < len) ? (len - 1 - t) : 0) : t;
      const float* xp = x + (st * NB + row) * INF_;
      float acc = bia[col];
      for (int k = 0; k < INF_; ++k) acc += xp[k] * Wih[col * INF_ + k];
      for (int k = 0; k < HH;  ++k) acc += hs[row][k] * Whh[col * HH + k];
      gs[row][col] = acc;
    }
    __syncthreads();
    for (int i = threadIdx.x; i < NB * HH; i += 1024) {
      int m = i >> 8, hc = i & 255;
      bool mk = (t < lens[m]);
      float gi = gs[m][hc], gf = gs[m][HH + hc], gg = gs[m][2 * HH + hc], go = gs[m][3 * HH + hc];
      float c2 = sigmf(gf) * cs[m][hc] + sigmf(gi) * tanhf(gg);
      float h2 = sigmf(go) * tanhf(c2);
      if (mk) { cs[m][hc] = c2; hs[m][hc] = h2; }
      float ov = mk ? h2 : 0.f;
      if (dir == 0) bi_out[(t * NB + m) * D2H + hc] = ov;
      else          out_r[(t * NB + m) * HH + hc]  = ov;
    }
    __syncthreads();
  }
}

// stage 2b/2c: assemble bi_out back-half, xbmat, revmat
__global__ void assembleA_kernel(const float* __restrict__ out_r, const int* __restrict__ lens,
                                 float* __restrict__ bi_out, float* __restrict__ xbmat) {
  int b = blockIdx.x >> 6, t = blockIdx.x & 63;
  bool mk = (t < lens[b]);
  int idx = mk ? (lens[b] - 1 - t) : 0;
  int c = threadIdx.x;             // 256 threads
  bi_out[(t * NB + b) * D2H + HH + c] = mk ? out_r[(idx * NB + b) * HH + c] : 0.f;
  __syncthreads();
  for (int cc = c; cc < D2H; cc += 256)
    xbmat[(b * ML + t) * D2H + cc] = bi_out[(t * NB + b) * D2H + cc];
}
__global__ void assembleB_kernel(const float* __restrict__ bi_out, const int* __restrict__ lens,
                                 float* __restrict__ revmat) {
  int b = blockIdx.x >> 6, t = blockIdx.x & 63;
  bool mk = (t < lens[b]);
  int idx = mk ? (lens[b] - 1 - t) : 0;
  for (int c = threadIdx.x; c < D2H; c += 256)
    revmat[(b * ML + t) * D2H + c] = mk ? bi_out[(idx * NB + b) * D2H + c] : 0.f;
}

// ---------------- stage 3: WMMA input projections C = A @ W.T + bias ----------------
__global__ void gemm_pre_kernel(const float* __restrict__ A, const _Float16* __restrict__ W16,
                                const float* __restrict__ bias, float* __restrict__ C) {
  __shared__ _Float16 As[64][64];
  __shared__ _Float16 Bs[64][64];
  const int K = D2H;
  int mblk = blockIdx.y * 64, nblk = blockIdx.x * 64;
  int wave = threadIdx.x >> 5, lane = threadIdx.x & 31;
  int mt = wave >> 2, nt = wave & 3;
  v8f acc = {};
  for (int k0 = 0; k0 < K; k0 += 64) {
    for (int i = threadIdx.x; i < 64 * 64; i += 512) {
      int r = i >> 6, kk = i & 63;
      As[r][kk] = (_Float16)A[(mblk + r) * K + k0 + kk];
      Bs[r][kk] = W16[(nblk + r) * K + k0 + kk];
    }
    __syncthreads();
#pragma unroll
    for (int ks = 0; ks < 2; ++ks) {
      v16h a  = load_a_frag(&As[0][0], 64, mt * 16, ks * 32);
      v16h bf = load_b_frag(&Bs[0][0], 64, nt * 16, ks * 32);
      acc = __builtin_amdgcn_wmma_f32_16x16x32_f16(false, a, false, bf, (short)0, acc, false, false);
    }
    __syncthreads();
  }
  int n  = nblk + nt * 16 + (lane & 15);
  int mb = mblk + mt * 16 + ((lane < 16) ? 0 : 8);
#pragma unroll
  for (int v = 0; v < 8; ++v) C[(mb + v) * G4 + n] = acc[v] + bias[n];
}

// ---------------- stage 4: attention-LSTM recurrence ----------------
// 1024 threads = 32 waves; wave w owns gate columns [w*32, w*32+32) with
// register-resident Whh fragments (no global weight traffic in the recurrence).
// Per-step `pre` row is double-buffered in LDS via async global->LDS DMA.
__global__ void att_lstm_kernel(const float* __restrict__ pre0, const float* __restrict__ preSeq,
                                const _Float16* __restrict__ Whh16, const int* __restrict__ lens,
                                float* __restrict__ hOut) {
  __shared__ _Float16 hs[16][HH];
  __shared__ float    cs[16][HH];
  __shared__ float    gs[16][G4];
  __shared__ float    pres[2][G4];
  int b = blockIdx.x >> 2, g = blockIdx.x & 3;
  int lenb = lens[b];
  int wave = threadIdx.x >> 5, lane = threadIdx.x & 31;

  // preload Whh fragments for this wave's two 16-col tiles (K=256 -> 8 k-steps)
  v16h wfrag[2][8];
#pragma unroll
  for (int q = 0; q < 2; ++q) {
    int n  = (wave * 2 + q) * 16 + (lane & 15);
    int kb = (lane < 16) ? 0 : 16;
#pragma unroll
    for (int kk = 0; kk < 8; ++kk) {
      v16h bf;
#pragma unroll
      for (int j = 0; j < 16; ++j) bf[j] = Whh16[n * HH + kk * 32 + kb + j];
      wfrag[q][kk] = bf;
    }
  }
  for (int i = threadIdx.x; i < 16 * HH; i += 1024) { hs[i >> 8][i & 255] = (_Float16)0.f; cs[i >> 8][i & 255] = 0.f; }
  __syncthreads();

  int Tg = (g * 16 >= lenb) ? 1 : ((g * 16 + 16 < lenb) ? (g * 16 + 16) : lenb);
  for (int t = 0; t < Tg; ++t) {
    // async-prefetch pre row for step t+1 (shared by all 16 rows) while we compute
    if (wave == 0 && (t + 1) < Tg)
      async_copy_wave(preSeq + (size_t)(b * ML + t) * G4, &pres[(t + 1) & 1][0], G4 * 4);

    v8f z = {};
    v8f acc[2] = {z, z};
#pragma unroll
    for (int kk = 0; kk < 8; ++kk) {                 // K = 256
      v16h a = load_a_frag(&hs[0][0], HH, 0, kk * 32);
      acc[0] = __builtin_amdgcn_wmma_f32_16x16x32_f16(false, a, false, wfrag[0][kk], (short)0, acc[0], false, false);
      acc[1] = __builtin_amdgcn_wmma_f32_16x16x32_f16(false, a, false, wfrag[1][kk], (short)0, acc[1], false, false);
    }
    const float* p0 = pre0 + (size_t)(b * ML + g * 16) * G4;   // t==0: per-row
#pragma unroll
    for (int q = 0; q < 2; ++q) {
      int n  = (wave * 2 + q) * 16 + (lane & 15);
      int mb = (lane < 16) ? 0 : 8;
#pragma unroll
      for (int v = 0; v < 8; ++v) {
        int m = mb + v;
        float p = (t == 0) ? p0[m * G4 + n] : pres[t & 1][n];
        gs[m][n] = acc[q][v] + p;
      }
    }
    __syncthreads();
    for (int i = threadIdx.x; i < 16 * HH; i += 1024) {
      int m = i >> 8, hc = i & 255;
      int ii = g * 16 + m;
      int rl = (ii < lenb) ? (ii + 1) : 1;
      if (t < rl) {
        float gi = gs[m][hc], gf = gs[m][HH + hc], gg = gs[m][2 * HH + hc], go = gs[m][3 * HH + hc];
        float c2 = sigmf(gf) * cs[m][hc] + sigmf(gi) * tanhf(gg);
        float h2 = sigmf(go) * tanhf(c2);
        cs[m][hc] = c2;
        hs[m][hc] = (_Float16)h2;
      }
    }
    if (wave == 0) wait_async0();   // drain DMA for next step's pre row
    __syncthreads();
  }
  for (int i = threadIdx.x; i < 16 * HH; i += 1024) {
    int m = i >> 8, hc = i & 255;
    hOut[(b * ML + g * 16 + m) * HH + hc] = (float)hs[m][hc];
  }
}

// ---------------- stage 5: h0b16 (B,65,1024) f16 ----------------
__global__ void assembleH0_kernel(const float* __restrict__ bi_out, const float* __restrict__ hL,
                                  const float* __restrict__ hR, _Float16* __restrict__ h0b16) {
  int b = blockIdx.x / (ML + 1), j = blockIdx.x % (ML + 1);
  for (int c = threadIdx.x; c < WRS_; c += 256) {
    float v = 0.f;
    if (j > 0) {
      int t = j - 1;
      if      (c < D2H)       v = bi_out[(t * NB + b) * D2H + c];
      else if (c < D2H + HH)  v = hL[(b * ML + t) * HH + (c - D2H)];
      else                    v = hR[(b * ML + t) * HH + (c - D2H - HH)];
    }
    h0b16[((size_t)(b * (ML + 1) + j)) * WRS_ + c] = (_Float16)v;
  }
}

__global__ void prefix_act_kernel(const int* __restrict__ act, int* __restrict__ cum) {
  if (threadIdx.x == 0) { int s = 0; for (int b = 0; b < NB; ++b) { cum[b] = s; s += act[b]; } cum[NB] = s; }
}

// ---------------- stage 6: classifier GEMM1 (gathered A) + ReLU ----------------
__global__ void gemm_cls1_kernel(const _Float16* __restrict__ h0b16, const int* __restrict__ sb,
                                 const float* __restrict__ W1, const float* __restrict__ b1,
                                 const int* __restrict__ act, _Float16* __restrict__ hcls) {
  int mblk = blockIdx.y * 64, nblk = blockIdx.x * 64;
  int bb0 = mblk >> 7, a0 = mblk & 127;
  if (a0 >= act[bb0]) return;                         // whole 64-row block unselected
  __shared__ _Float16 As[64][64];
  __shared__ _Float16 Bs[64][64];
  int wave = threadIdx.x >> 5, lane = threadIdx.x & 31;
  int mt = wave >> 5 ? 0 : (wave >> 2), nt = wave & 3;
  mt = wave >> 2;
  v8f acc = {};
  for (int k0 = 0; k0 < K1; k0 += 64) {
    for (int i = threadIdx.x; i < 64 * 64; i += 512) {
      int r = i >> 6, kk = i & 63;
      int k = k0 + kk;
      int row = mblk + r;
      int bb = row >> 7, aa = row & 127;
      int e = k >> 10, cc = k & 1023;
      int j = sb[(bb * MAL + aa) * ELE + e] + 1;
      As[r][kk] = h0b16[((size_t)(bb * (ML + 1) + j)) * WRS_ + cc];
      Bs[r][kk] = (_Float16)W1[(size_t)(nblk + r) * K1 + k];
      if (k + 64 < K1) __builtin_prefetch(&W1[(size_t)(nblk + r) * K1 + k + 64], 0, 1);
    }
    __syncthreads();
#pragma unroll
    for (int ks = 0; ks < 2; ++ks) {
      v16h a  = load_a_frag(&As[0][0], 64, mt * 16, ks * 32);
      v16h bf = load_b_frag(&Bs[0][0], 64, nt * 16, ks * 32);
      acc = __builtin_amdgcn_wmma_f32_16x16x32_f16(false, a, false, bf, (short)0, acc, false, false);
    }
    __syncthreads();
  }
  int n  = nblk + nt * 16 + (lane & 15);
  int mb = mblk + mt * 16 + ((lane < 16) ? 0 : 8);
#pragma unroll
  for (int v = 0; v < 8; ++v) {
    float xv = acc[v] + b1[n];
    hcls[(size_t)(mb + v) * N1 + n] = (_Float16)(xv > 0.f ? xv : 0.f);
  }
}

// ---------------- stage 7: GEMM2 -> compacted output ----------------
__global__ void gemm_cls2_kernel(const _Float16* __restrict__ hcls, const _Float16* __restrict__ w2p,
                                 const float* __restrict__ b2, const int* __restrict__ act,
                                 const int* __restrict__ cum, float* __restrict__ out) {
  int r0 = blockIdx.x * 16;
  int b = r0 >> 7, a0 = r0 & 127;
  if (a0 >= act[b]) return;
  __shared__ _Float16 As[16][64];
  int wave = threadIdx.x >> 5, lane = threadIdx.x & 31;   // 8 waves: N tiles 0..7 (padded W2)
  v8f acc = {};
  for (int k0 = 0; k0 < N1; k0 += 64) {
    for (int i = threadIdx.x; i < 16 * 64; i += 256) {
      int r = i >> 6, kk = i & 63;
      As[r][kk] = hcls[(size_t)(r0 + r) * N1 + k0 + kk];
    }
    __syncthreads();
#pragma unroll
    for (int ks = 0; ks < 2; ++ks) {
      v16h a = load_a_frag(&As[0][0], 64, 0, ks * 32);
      int n  = wave * 16 + (lane & 15);
      int kb = (lane < 16) ? 0 : 16;
      v16h bf;
#pragma unroll
      for (int j = 0; j < 16; ++j) bf[j] = w2p[(size_t)n * N1 + k0 + ks * 32 + kb + j];
      acc = __builtin_amdgcn_wmma_f32_16x16x32_f16(false, a, false, bf, (short)0, acc, false, false);
    }
    __syncthreads();
  }
  int n = wave * 16 + (lane & 15);
  if (n < NCAT) {
    int mb = (lane < 16) ? 0 : 8;
#pragma unroll
    for (int v = 0; v < 8; ++v) {
      int row = r0 + mb + v;
      int aa = row & 127;
      if (aa < act[b]) out[(size_t)(cum[b] + aa) * NCAT + n] = acc[v] + b2[n];
    }
  }
}

// =======================================================================
extern "C" void kernel_launch(void* const* d_in, const int* in_sizes, int n_in,
                              void* d_out, int out_size, void* d_ws, size_t ws_size,
                              hipStream_t stream) {
  (void)in_sizes; (void)n_in; (void)out_size; (void)ws_size;
  enum { I_WEMB = 0, I_PEMB, I_BF_WIH, I_BF_WHH, I_BF_B, I_BB_WIH, I_BB_WHH, I_BB_B,
         I_AL_WIH, I_AL_WHH, I_AL_B, I_AR_WIH, I_AR_WHH, I_AR_B,
         I_W1, I_B1, I_W2, I_B2, I_SB, I_DIC, I_POS, I_SLEN, I_ALEN };

  const float* wemb = (const float*)d_in[I_WEMB];
  const float* pemb = (const float*)d_in[I_PEMB];
  const float* bfWih = (const float*)d_in[I_BF_WIH];
  const float* bfWhh = (const float*)d_in[I_BF_WHH];
  const float* bfB   = (const float*)d_in[I_BF_B];
  const float* bbWih = (const float*)d_in[I_BB_WIH];
  const float* bbWhh = (const float*)d_in[I_BB_WHH];
  const float* bbB   = (const float*)d_in[I_BB_B];
  const float* alWih = (const float*)d_in[I_AL_WIH];
  const float* alWhh = (const float*)d_in[I_AL_WHH];
  const float* alB   = (const float*)d_in[I_AL_B];
  const float* arWih = (const float*)d_in[I_AR_WIH];
  const float* arWhh = (const float*)d_in[I_AR_WHH];
  const float* arB   = (const float*)d_in[I_AR_B];
  const float* W1    = (const float*)d_in[I_W1];
  const float* b1    = (const float*)d_in[I_B1];
  const float* W2    = (const float*)d_in[I_W2];
  const float* b2    = (const float*)d_in[I_B2];
  const int*   sb    = (const int*)d_in[I_SB];
  const int*   dic   = (const int*)d_in[I_DIC];
  const int*   pos   = (const int*)d_in[I_POS];
  const int*   slen  = (const int*)d_in[I_SLEN];
  const int*   alen  = (const int*)d_in[I_ALEN];
  float* out = (float*)d_out;

  // ---- workspace carve-up (all 256B aligned) ----
  char* ws = (char*)d_ws;
  size_t off = 0;
  auto carve = [&](size_t bytes) { char* p = ws + off; off += (bytes + 255) & ~(size_t)255; return p; };
  float*     x       = (float*)carve((size_t)ML * NB * INF_ * 4);
  float*     out_r   = (float*)carve((size_t)ML * NB * HH * 4);
  float*     bi_out  = (float*)carve((size_t)ML * NB * D2H * 4);
  float*     xbmat   = (float*)carve((size_t)NB * ML * D2H * 4);
  float*     revmat  = (float*)carve((size_t)NB * ML * D2H * 4);
  float*     preL    = (float*)carve((size_t)NB * ML * G4 * 4);
  float*     preR0   = (float*)carve((size_t)NB * ML * G4 * 4);
  float*     preRs   = (float*)carve((size_t)NB * ML * G4 * 4);
  float*     hL      = (float*)carve((size_t)NB * ML * HH * 4);
  float*     hR      = (float*)carve((size_t)NB * ML * HH * 4);
  _Float16*  h0b16   = (_Float16*)carve((size_t)NB * (ML + 1) * WRS_ * 2);
  _Float16*  hcls16  = (_Float16*)carve((size_t)NB * MAL * N1 * 2);
  _Float16*  wihl16  = (_Float16*)carve((size_t)G4 * D2H * 2);
  _Float16*  wihr16  = (_Float16*)carve((size_t)G4 * D2H * 2);
  _Float16*  whhl16  = (_Float16*)carve((size_t)G4 * HH * 2);
  _Float16*  whhr16  = (_Float16*)carve((size_t)G4 * HH * 2);
  _Float16*  w2p16   = (_Float16*)carve((size_t)128 * N1 * 2);
  int*       cum     = (int*)carve(32 * 4);

  // 0) weight conversions
  cvt16_kernel<<<(G4 * D2H + 255) / 256, 256, 0, stream>>>(alWih, wihl16, G4 * D2H);
  cvt16_kernel<<<(G4 * D2H + 255) / 256, 256, 0, stream>>>(arWih, wihr16, G4 * D2H);
  cvt16_kernel<<<(G4 * HH + 255) / 256, 256, 0, stream>>>(alWhh, whhl16, G4 * HH);
  cvt16_kernel<<<(G4 * HH + 255) / 256, 256, 0, stream>>>(arWhh, whhr16, G4 * HH);
  cvtW2_kernel<<<(128 * N1 + 255) / 256, 256, 0, stream>>>(W2, w2p16);

  // 1) embeddings
  embed_kernel<<<NB * ML, 160, 0, stream>>>(wemb, pemb, dic, pos, x);

  // 2) bi-LSTM + assembly
  bilstm_kernel<<<2, 1024, 0, stream>>>(x, bfWih, bfWhh, bfB, bbWih, bbWhh, bbB, slen, bi_out, out_r);
  assembleA_kernel<<<NB * ML, 256, 0, stream>>>(out_r, slen, bi_out, xbmat);
  assembleB_kernel<<<NB * ML, 256, 0, stream>>>(bi_out, slen, revmat);

  // 3) input projections (WMMA GEMMs)
  dim3 gpre(G4 / 64, (NB * ML) / 64);
  gemm_pre_kernel<<<gpre, 512, 0, stream>>>(xbmat, wihl16, alB, preL);
  gemm_pre_kernel<<<gpre, 512, 0, stream>>>(xbmat, wihr16, arB, preR0);
  gemm_pre_kernel<<<gpre, 512, 0, stream>>>(revmat, wihr16, arB, preRs);

  // 4) attention LSTM recurrences (register-resident Whh WMMA + async pre DMA)
  att_lstm_kernel<<<NB * 4, 1024, 0, stream>>>(preL, preL, whhl16, slen, hL);
  att_lstm_kernel<<<NB * 4, 1024, 0, stream>>>(preR0, preRs, whhr16, slen, hR);

  // 5) h0b (f16) + selection prefix sums
  assembleH0_kernel<<<NB * (ML + 1), 256, 0, stream>>>(bi_out, hL, hR, h0b16);
  prefix_act_kernel<<<1, 32, 0, stream>>>(alen, cum);

  // 6) classifier GEMM1 + ReLU (gathered A, WMMA)
  dim3 g1(N1 / 64, (NB * MAL) / 64);
  gemm_cls1_kernel<<<g1, 512, 0, stream>>>(h0b16, sb, W1, b1, alen, hcls16);

  // 7) GEMM2 -> compacted logits
  gemm_cls2_kernel<<<(NB * MAL) / 16, 256, 0, stream>>>(hcls16, w2p16, b2, alen, cum, out);
}
